// EdgeTransition_10479720203131
// MI455X (gfx1250) — matrix operand order repaired and checked by reading, not compiled, not run on hardware
//
#include <hip/hip_runtime.h>
#include <hip/hip_bf16.h>

// ---------------------------------------------------------------------------
// CDNA5 (gfx1250) fused EdgeTransition:
//   nh  = node_emb @ W0 + b0                   (bf16 WMMA, f32 accum)
//   h   = [edge_emb | nh[src] | nh[dst]]       (TDM + async gather into LDS)
//   t1  = relu(h  @ Wt1 + bt1)                 (WMMA, LDS->LDS)
//   t2  = relu(t1 @ Wt2 + bt2)                 (WMMA, LDS->LDS)
//   r   = t2 + h                               (v_pk_add_bf16 residual in LDS)
//   out = LayerNorm(r @ Wf + bf) * gamma + beta
// Weights pre-transposed+converted to bf16 W^T[n][k] (L2-resident, ~0.75MB).
// Each wave owns a 48-col N slice over all 64 rows -> B fragments loaded
// exactly once per block. fp32->bf16 pair conversion = one v_perm_b32.
// ---------------------------------------------------------------------------

typedef __attribute__((ext_vector_type(16))) __bf16 v16bf;
typedef __attribute__((ext_vector_type(8)))  float  v8f;
typedef __attribute__((ext_vector_type(4)))  unsigned int u32x4;
typedef __attribute__((ext_vector_type(8)))  int i32x8;
typedef __attribute__((ext_vector_type(4)))  int i32x4;
typedef int v4i __attribute__((vector_size(16)));   // async builtin pointee type

#define N_NODES 10000
#define N_EDGES 200000
#define D_NODE  256
#define D_BIAS  128
#define D_EIN   128
#define D_HID   384
#define D_EOUT  128

#define BM      64            // edges per block
#define PITCH   392           // D_HID + 8 pad (multiple of 8 elems = 16B)
#define OPITCH  132           // fp32 out tile pitch

// workspace offsets (bytes)
#define OFF_W0T 0u
#define OFF_W1T 65536u
#define OFF_W2T 360448u
#define OFF_WFT 655360u
#define OFF_NH  753664u

#if defined(__gfx1250__) && __has_builtin(__builtin_amdgcn_tensor_load_to_lds)
#define HAS_TDM 1
#endif
#if defined(__gfx1250__) && __has_builtin(__builtin_amdgcn_global_load_async_to_lds_b128)
#define HAS_ASYNC 1
#endif

#define AS1 __attribute__((address_space(1)))
#define AS3 __attribute__((address_space(3)))

// ---- bf16 helpers ----------------------------------------------------------
__device__ __forceinline__ unsigned fbits(float f) {
  union { float f; unsigned u; } c; c.f = f; return c.u;
}
// RNE scalar convert (weights only)
__device__ __forceinline__ unsigned short f2bf(float f) {
  unsigned u = fbits(f);
  unsigned r = u + 0x7FFFu + ((u >> 16) & 1u);
  return (unsigned short)(r >> 16);
}
// truncating scalar convert: folds into *_d16_hi stores
__device__ __forceinline__ unsigned short f2bf_t(float f) {
  return (unsigned short)(fbits(f) >> 16);
}
// truncating pair convert: one v_perm_b32 (lo halfword <- a, hi <- b)
__device__ __forceinline__ unsigned pk2bf_t(float a, float b) {
  return __builtin_amdgcn_perm(fbits(b), fbits(a), 0x07060302u);
}
// packed bf16 add (VOP3P V_PK_ADD_BF16)
__device__ __forceinline__ unsigned pk_add_bf16(unsigned a, unsigned b) {
#if defined(__gfx1250__)
  unsigned d;
  asm("v_pk_add_bf16 %0, %1, %2" : "=v"(d) : "v"(a), "v"(b));
  return d;
#else
  return 0u;
#endif
}

union FragU { uint4 q[2]; v16bf v; unsigned short h[16]; unsigned d[8]; };

// A fragment (16x32 bf16) from row-major bf16 LDS tile (ISA 7.12.2 layout).
__device__ __forceinline__ v16bf lds_afrag(const unsigned short* base, int m,
                                           int pitch, int k0, int half) {
  const unsigned short* p = base + m * pitch + k0 + half * 8;
  FragU u;
  u.q[0] = *(const uint4*)(p);
  u.q[1] = *(const uint4*)(p + 16);
  return u.v;
}

// B fragment (32x16 bf16) from W^T bf16 [N][K] in global.
__device__ __forceinline__ v16bf glb_bfrag(const unsigned short* WT, int n,
                                           int K, int k0, int half) {
  const unsigned short* p = WT + (size_t)n * K + k0 + half * 8;
  FragU u;
  u.q[0] = *(const uint4*)(p);
  u.q[1] = *(const uint4*)(p + 16);
  return u.v;
}

// A fragment converted on the fly from row-major fp32 global (perm pack).
__device__ __forceinline__ v16bf glb_afrag_f32(const float* A, int m, int ldk,
                                               int k0, int half) {
  const float* p = A + (size_t)m * ldk + k0 + half * 8;
  FragU u;
#pragma unroll
  for (int j = 0; j < 4; ++j) u.d[j]     = pk2bf_t(p[2 * j],      p[2 * j + 1]);
#pragma unroll
  for (int j = 0; j < 4; ++j) u.d[4 + j] = pk2bf_t(p[16 + 2 * j], p[16 + 2 * j + 1]);
  return u.v;
}

// ---- weight transpose + bf16 convert: in fp32 [K][N] -> out bf16 [N][K] ----
__global__ void wt_transpose_bf16(const float* __restrict__ in,
                                  unsigned short* __restrict__ out,
                                  int K, int N) {
  int idx = blockIdx.x * 256 + threadIdx.x;
  if (idx >= K * N) return;
  int k = idx / N, n = idx - k * N;
  out[(size_t)n * K + k] = f2bf(in[idx]);
}

// ---- node GEMM: nh = node_emb @ W0 + b0 -> bf16 [N_NODES][128] -------------
__global__ __launch_bounds__(256) void node_gemm(
    const float* __restrict__ node_emb, const unsigned short* __restrict__ W0T,
    const float* __restrict__ b0, unsigned short* __restrict__ nh) {
  const int lane = threadIdx.x & 31;
  const int wave = threadIdx.x >> 5;      // 8 waves -> 8 N tiles
  const int half = lane >> 4;
  const int ln   = lane & 15;
  const int mb   = blockIdx.x * 16;       // 625 blocks, exact
  const int n0   = wave * 16;

  v8f acc = {};
#pragma unroll
  for (int kt = 0; kt < 8; ++kt) {        // K = 256
    v16bf a = glb_afrag_f32(node_emb, mb + ln, D_NODE, kt * 32, half);
    v16bf b = glb_bfrag(W0T, n0 + ln, D_NODE, kt * 32, half);
    acc = __builtin_amdgcn_wmma_f32_16x16x32_bf16(false, a, false, b,
                                                  (short)0, acc, false, false);
  }
  const float bv = b0[n0 + ln];
#pragma unroll
  for (int r = 0; r < 8; ++r) {
    int m = mb + r + half * 8;
    nh[(size_t)m * 128 + n0 + ln] = f2bf_t(acc[r] + bv);
  }
}

// ---- trunk layer: dst = act(src @ W^T + b) -------------------------------
// Wave w owns N cols [w*48, w*48+48): B fragments cached in registers and
// reused across all 4 M stripes -> each weight byte read once per block.
template <bool RELU>
__device__ __forceinline__ void gemm_layer(const unsigned short* __restrict__ srcS,
                                           unsigned short* __restrict__ dstS,
                                           const unsigned short* __restrict__ WT,
                                           const float* __restrict__ bias,
                                           int lane, int wave) {
  const int half = lane >> 4;
  const int ln   = lane & 15;
#pragma unroll
  for (int j = 0; j < 3; ++j) {
    const int n0 = wave * 48 + j * 16;
    v16bf b[12];
#pragma unroll
    for (int kt = 0; kt < 12; ++kt)
      b[kt] = glb_bfrag(WT, n0 + ln, D_HID, kt * 32, half);
    const float bv = bias[n0 + ln];
#pragma unroll
    for (int ms = 0; ms < 4; ++ms) {
      const int mbase = ms * 16;
      v8f acc = {};
#pragma unroll
      for (int kt = 0; kt < 12; ++kt) {
        v16bf a = lds_afrag(srcS, mbase + ln, PITCH, kt * 32, half);
        acc = __builtin_amdgcn_wmma_f32_16x16x32_bf16(false, a, false, b[kt],
                                                      (short)0, acc, false, false);
      }
#pragma unroll
      for (int r = 0; r < 8; ++r) {
        float v = acc[r] + bv;
        if (RELU) v = fmaxf(v, 0.0f);
        dstS[(mbase + r + half * 8) * PITCH + n0 + ln] = f2bf_t(v);
      }
    }
  }
}

// ---- fused edge kernel -----------------------------------------------------
__global__ __launch_bounds__(256) void edge_fused(
    const float* __restrict__ edge_emb, const int* __restrict__ edge_index,
    const unsigned short* __restrict__ nh,
    const unsigned short* __restrict__ W1T, const float* __restrict__ bt1,
    const unsigned short* __restrict__ W2T, const float* __restrict__ bt2,
    const unsigned short* __restrict__ WfT, const float* __restrict__ bf_,
    const float* __restrict__ gamma, const float* __restrict__ beta,
    float* __restrict__ out) {
  __shared__ alignas(16) unsigned short hS[BM * PITCH];
  __shared__ alignas(16) unsigned short t1S[BM * PITCH];
  __shared__ alignas(16) unsigned short t2S[BM * PITCH];   // also TDM fp32 stage
  __shared__ float sumS[256];
  __shared__ float sqS[256];
  __shared__ float mS[BM];
  __shared__ float rstdS[BM];
  float* outS = (float*)hS;               // alias: h dead after residual

  const int tid  = threadIdx.x;
  const int lane = tid & 31;
  const int wave = tid >> 5;
  const int e0   = blockIdx.x * BM;       // 3125 blocks, exact

  // ---- stage edge_emb tile via Tensor Data Mover (fp32 [64][128] -> t2S) ----
#if HAS_TDM
  if (wave == 0) {
    unsigned lds_off = (unsigned)(uintptr_t)(void*)t2S;
    unsigned long long ga =
        (unsigned long long)(uintptr_t)(const void*)(edge_emb + (size_t)e0 * D_EIN);
    u32x4 g0 = { 0x1u,                                   // count=1
                 lds_off,                                // lds_addr
                 (unsigned)(ga & 0xFFFFFFFFu),           // global_addr lo
                 (unsigned)((ga >> 32) & 0x01FFFFFFu) | (2u << 30) }; // hi|type=2
    i32x8 g1 = { (int)(2u << 16),        // data_size = 4B
                 (int)(128u << 16),      // tensor_dim0[15:0] = 128
                 (int)(64u << 16),       // tensor_dim1[15:0] = 64
                 (int)(128u << 16),      // tile_dim0 = 128
                 (int)64,                // tile_dim1 = 64
                 (int)128,               // tensor_dim0_stride lo32 = 128
                 0, 0 };
    i32x4 z4 = {0, 0, 0, 0};
#if __clang_major__ >= 23
    i32x8 z8 = {0, 0, 0, 0, 0, 0, 0, 0};
    __builtin_amdgcn_tensor_load_to_lds(g0, g1, z4, z4, z8, 0);
#else
    __builtin_amdgcn_tensor_load_to_lds(g0, g1, z4, z4, 0);
#endif
  }
#else
  // fallback: direct load + convert edge_emb -> hS cols 0..127
#pragma unroll
  for (int t = 0; t < 8; ++t) {           // 64 rows x 32 float4
    int idx = tid + t * 256;
    int r = idx >> 5, c4 = idx & 31;
    float4 f = ((const float4*)(edge_emb + (size_t)(e0 + r) * D_EIN))[c4];
    unsigned* d = (unsigned*)(hS + r * PITCH) + c4 * 2;
    d[0] = pk2bf_t(f.x, f.y);
    d[1] = pk2bf_t(f.z, f.w);
  }
#endif

  // ---- gather nh[src], nh[dst] -> hS cols 128..383 (async to LDS) ----
#pragma unroll
  for (int t = 0; t < 4; ++t) {           // src: 64 rows x 16 uint4
    int idx = tid + t * 256;
    int r = idx >> 4, c = idx & 15;
    int node = edge_index[e0 + r];
#if HAS_ASYNC
    uintptr_t gp = (uintptr_t)(const void*)(nh + (size_t)node * 128) + (size_t)c * 16;
    uintptr_t lp = (uintptr_t)(unsigned)(uintptr_t)(const void*)(hS + r * PITCH + 128)
                 + (uintptr_t)(c * 16);
    __builtin_amdgcn_global_load_async_to_lds_b128((AS1 v4i*)gp, (AS3 v4i*)lp, 0, 0);
#else
    ((uint4*)(hS + r * PITCH + 128))[c] =
        ((const uint4*)(nh + (size_t)node * 128))[c];
#endif
  }
#pragma unroll
  for (int t = 0; t < 4; ++t) {           // dst
    int idx = tid + t * 256;
    int r = idx >> 4, c = idx & 15;
    int node = edge_index[N_EDGES + e0 + r];
#if HAS_ASYNC
    uintptr_t gp = (uintptr_t)(const void*)(nh + (size_t)node * 128) + (size_t)c * 16;
    uintptr_t lp = (uintptr_t)(unsigned)(uintptr_t)(const void*)(hS + r * PITCH + 256)
                 + (uintptr_t)(c * 16);
    __builtin_amdgcn_global_load_async_to_lds_b128((AS1 v4i*)gp, (AS3 v4i*)lp, 0, 0);
#else
    ((uint4*)(hS + r * PITCH + 256))[c] =
        ((const uint4*)(nh + (size_t)node * 128))[c];
#endif
  }

#if HAS_ASYNC
#if __has_builtin(__builtin_amdgcn_s_wait_asynccnt)
  __builtin_amdgcn_s_wait_asynccnt(0);
#else
  asm volatile("s_wait_asynccnt 0x0" ::: "memory");
#endif
#endif
#if HAS_TDM
  if (wave == 0) __builtin_amdgcn_s_wait_tensorcnt(0);
#endif
  __syncthreads();

#if HAS_TDM
  // convert staged fp32 tile (t2S) -> hS cols 0..127 bf16 (perm pack)
  {
    const float* stage = (const float*)t2S;
#pragma unroll
    for (int t = 0; t < 8; ++t) {
      int idx = tid + t * 256;
      int r = idx >> 5, c4 = idx & 31;
      float4 f = ((const float4*)(stage + r * D_EIN))[c4];
      unsigned* d = (unsigned*)(hS + r * PITCH) + c4 * 2;
      d[0] = pk2bf_t(f.x, f.y);
      d[1] = pk2bf_t(f.z, f.w);
    }
  }
  __syncthreads();
#endif

  // ---- trunk ----
  gemm_layer<true>(hS, t1S, W1T, bt1, lane, wave);
  __syncthreads();
  gemm_layer<true>(t1S, t2S, W2T, bt2, lane, wave);
  __syncthreads();

  // ---- residual: t2 += h (one v_pk_add_bf16 per 2 elems) ----
  for (int i = tid; i < (BM * PITCH) / 2; i += 256) {
    ((unsigned*)t2S)[i] = pk_add_bf16(((unsigned*)t2S)[i], ((unsigned*)hS)[i]);
  }
  __syncthreads();

  // ---- final GEMM: outS = r @ Wf^T + bf (fp32, N=128); wave owns 1 N tile --
  {
    const int half = lane >> 4;
    const int ln   = lane & 15;
    const int n0   = wave * 16;
    v16bf b[12];
#pragma unroll
    for (int kt = 0; kt < 12; ++kt)
      b[kt] = glb_bfrag(WfT, n0 + ln, D_HID, kt * 32, half);
    const float bv = bf_[n0 + ln];
#pragma unroll
    for (int ms = 0; ms < 4; ++ms) {
      const int mbase = ms * 16;
      v8f acc = {};
#pragma unroll
      for (int kt = 0; kt < 12; ++kt) {
        v16bf a = lds_afrag(t2S, mbase + ln, PITCH, kt * 32, half);
        acc = __builtin_amdgcn_wmma_f32_16x16x32_bf16(false, a, false, b[kt],
                                                      (short)0, acc, false, false);
      }
#pragma unroll
      for (int r = 0; r < 8; ++r)
        outS[(mbase + r + half * 8) * OPITCH + n0 + ln] = acc[r] + bv;
    }
  }
  __syncthreads();

  // ---- LayerNorm over 128 cols, 4 threads per row ----
  {
    const int q = tid & 3;
    const int r = tid >> 2;
    const float* row = outS + r * OPITCH + q * 32;
    float s = 0.f, ss = 0.f;
#pragma unroll
    for (int c = 0; c < 32; ++c) { float v = row[c]; s += v; ss += v * v; }
    sumS[tid] = s; sqS[tid] = ss;
    __syncthreads();
    if (q == 0) {
      float S  = sumS[r * 4] + sumS[r * 4 + 1] + sumS[r * 4 + 2] + sumS[r * 4 + 3];
      float SS = sqS[r * 4]  + sqS[r * 4 + 1]  + sqS[r * 4 + 2]  + sqS[r * 4 + 3];
      float mean = S * (1.0f / 128.0f);
      float var  = SS * (1.0f / 128.0f) - mean * mean;
      mS[r] = mean;
      rstdS[r] = rsqrtf(var + 1e-5f);
    }
    __syncthreads();
    const float mean = mS[r], rstd = rstdS[r];
    float* orow = out + (size_t)(e0 + r) * D_EOUT + q * 32;
#pragma unroll
    for (int c = 0; c < 32; ++c) {
      int col = q * 32 + c;
      orow[c] = (row[c] - mean) * rstd * gamma[col] + beta[col];
    }
  }
}

// ---------------------------------------------------------------------------
extern "C" void kernel_launch(void* const* d_in, const int* in_sizes, int n_in,
                              void* d_out, int out_size, void* d_ws, size_t ws_size,
                              hipStream_t stream) {
  const float* node_emb   = (const float*)d_in[0];
  const float* edge_emb   = (const float*)d_in[1];
  const int*   edge_index = (const int*)d_in[2];
  const float* W0  = (const float*)d_in[3];
  const float* b0  = (const float*)d_in[4];
  const float* Wt1 = (const float*)d_in[5];
  const float* bt1 = (const float*)d_in[6];
  const float* Wt2 = (const float*)d_in[7];
  const float* bt2 = (const float*)d_in[8];
  const float* Wf  = (const float*)d_in[9];
  const float* bf_ = (const float*)d_in[10];
  const float* gamma = (const float*)d_in[11];
  const float* beta  = (const float*)d_in[12];
  float* out = (float*)d_out;

  unsigned char* ws = (unsigned char*)d_ws;
  unsigned short* W0T = (unsigned short*)(ws + OFF_W0T);
  unsigned short* W1T = (unsigned short*)(ws + OFF_W1T);
  unsigned short* W2T = (unsigned short*)(ws + OFF_W2T);
  unsigned short* WfT = (unsigned short*)(ws + OFF_WFT);
  unsigned short* nh  = (unsigned short*)(ws + OFF_NH);

  wt_transpose_bf16<<<(D_NODE * D_BIAS + 255) / 256, 256, 0, stream>>>(W0, W0T, D_NODE, D_BIAS);
  wt_transpose_bf16<<<(D_HID * D_HID + 255) / 256, 256, 0, stream>>>(Wt1, W1T, D_HID, D_HID);
  wt_transpose_bf16<<<(D_HID * D_HID + 255) / 256, 256, 0, stream>>>(Wt2, W2T, D_HID, D_HID);
  wt_transpose_bf16<<<(D_HID * D_EOUT + 255) / 256, 256, 0, stream>>>(Wf, WfT, D_HID, D_EOUT);

  node_gemm<<<N_NODES / 16, 256, 0, stream>>>(node_emb, W0T, b0, nh);

  edge_fused<<<N_EDGES / BM, 256, 0, stream>>>(edge_emb, edge_index, nh,
                                               W1T, bt1, W2T, bt2, WfT, bf_,
                                               gamma, beta, out);
}
